// SREXmodel_76785425318724
// MI455X (gfx1250) — matrix operand order repaired and checked by reading, not compiled, not run on hardware
//
#include <hip/hip_runtime.h>
#include <hip/hip_bf16.h>
#include <stdint.h>

typedef __attribute__((ext_vector_type(16))) _Float16 v16h;
typedef __attribute__((ext_vector_type(8)))  _Float16 v8h;
typedef __attribute__((ext_vector_type(8)))  float    v8f;

#define NCLI 30000
#define EPE  30000
#define EFE  240000
#define HH   8
#define KB   128   // K-block staged per LDS buffer (4 WMMA K-steps)

// ---------- ordered-float encoding for atomic max ----------
__device__ __forceinline__ unsigned fenc(float f) {
  unsigned u = __float_as_uint(f);
  return (u & 0x80000000u) ? ~u : (u | 0x80000000u);
}
__device__ __forceinline__ float fdec(unsigned u) {
  unsigned b = (u & 0x80000000u) ? (u & 0x7FFFFFFFu) : ~u;
  return __uint_as_float(b);
}

// ---------- utility kernels ----------
__global__ void k_fill_u32(unsigned* __restrict__ p, unsigned v, long n) {
  long i = (long)blockIdx.x * blockDim.x + threadIdx.x;
  if (i < n) p[i] = v;
}

// pad/convert rows: x[N,Kin] f32 -> xh[N,Kp] f16 (zero padded)
__global__ void k_pad_f16(const float* __restrict__ x, _Float16* __restrict__ xh,
                          long N, int Kin, int Kp) {
  long i = (long)blockIdx.x * blockDim.x + threadIdx.x;
  if (i >= N * Kp) return;
  int k = (int)(i % Kp);
  long n = i / Kp;
  xh[i] = (k < Kin) ? (_Float16)x[n * Kin + k] : (_Float16)0.0f;
}

// transpose+convert weights: W[K,Nn] f32 -> WT[Nn,Kp] f16, WT[n,k]=W[k,n]
__global__ void k_wT_f16(const float* __restrict__ Wm, _Float16* __restrict__ WT,
                         int K, int Nn, int Kp) {
  long i = (long)blockIdx.x * blockDim.x + threadIdx.x;
  if (i >= (long)Nn * Kp) return;
  int k = (int)(i % Kp);
  int nn = (int)(i / Kp);
  WT[i] = (k < K) ? (_Float16)Wm[(size_t)k * Nn + nn] : (_Float16)0.0f;
}

// ---------- simple WMMA GEMM (small K): one wave -> 16x64 tile ----------
__global__ __launch_bounds__(256) void k_wmma_gemm(
    const _Float16* __restrict__ Ah, const _Float16* __restrict__ Bt,
    float* __restrict__ C, int M, int Nt, int Kt) {
  int lane = threadIdx.x & 31;
  int gw = blockIdx.x * (blockDim.x >> 5) + (threadIdx.x >> 5);
  int tilesN = Nt >> 6;
  int tM = gw / tilesN;
  int tN = gw % tilesN;
  if (tM * 16 >= M) return;

  int row = tM * 16 + (lane & 15);
  int colBase = tN * 64 + (lane & 15);
  int kbA = (lane < 16) ? 0 : 16;
  int kbB = (lane < 16) ? 0 : 16;
  const _Float16* arow = Ah + (size_t)row * Kt;
  const _Float16* b0p = Bt + (size_t)(colBase + 0)  * Kt + kbB;
  const _Float16* b1p = Bt + (size_t)(colBase + 16) * Kt + kbB;
  const _Float16* b2p = Bt + (size_t)(colBase + 32) * Kt + kbB;
  const _Float16* b3p = Bt + (size_t)(colBase + 48) * Kt + kbB;

  v8f c0 = {}, c1 = {}, c2 = {}, c3 = {};
  for (int k0 = 0; k0 < Kt; k0 += 32) {
    v8h alo = *(const v8h*)(arow + k0 + kbA);
    v8h ahi = *(const v8h*)(arow + k0 + kbA + 8);
    v16h a;
#pragma unroll
    for (int i = 0; i < 8; ++i) { a[i] = alo[i]; a[8 + i] = ahi[i]; }
    v16h b0 = *(const v16h*)(b0p + k0);
    v16h b1 = *(const v16h*)(b1p + k0);
    v16h b2 = *(const v16h*)(b2p + k0);
    v16h b3 = *(const v16h*)(b3p + k0);
    c0 = __builtin_amdgcn_wmma_f32_16x16x32_f16(false, a, false, b0, (short)0, c0, false, false);
    c1 = __builtin_amdgcn_wmma_f32_16x16x32_f16(false, a, false, b1, (short)0, c1, false, false);
    c2 = __builtin_amdgcn_wmma_f32_16x16x32_f16(false, a, false, b2, (short)0, c2, false, false);
    c3 = __builtin_amdgcn_wmma_f32_16x16x32_f16(false, a, false, b3, (short)0, c3, false, false);
  }
  int mHi = (lane >= 16) ? 8 : 0;
  int nlo = lane & 15;
#pragma unroll
  for (int i = 0; i < 8; ++i) {
    size_t rbase = (size_t)(tM * 16 + i + mHi) * Nt + tN * 64;
    C[rbase + 0  + nlo] = c0[i];
    C[rbase + 16 + nlo] = c1[i];
    C[rbase + 32 + nlo] = c2[i];
    C[rbase + 48 + nlo] = c3[i];
  }
}

// ---------- LDS-pipelined WMMA GEMM (K multiple of 128) ----------
// Block = 8 waves; block macro-tile 32(M) x 512(N); wave tile 32(M) x 64(N).
// The 32-row A slab is staged to LDS with async global->LDS loads (double
// buffered, s_wait_asynccnt) and shared by all 8 waves.
__global__ __launch_bounds__(256) void k_wmma_gemm_lds(
    const _Float16* __restrict__ Ah, const _Float16* __restrict__ Bt,
    float* __restrict__ C, int M, int Nt, int Kt) {
  __shared__ _Float16 As[2 * 32 * KB];   // 2 x 8KB
  int lane = threadIdx.x & 31;
  int wv = threadIdx.x >> 5;
  int blocksN = Nt >> 9;
  int bM = blockIdx.x / blocksN;
  int bN = blockIdx.x % blocksN;
  int colBase = bN * 512 + wv * 64 + (lane & 15);
  int kbA = (lane < 16) ? 0 : 16;
  int kbB = (lane < 16) ? 0 : 16;
  const _Float16* b0p = Bt + (size_t)(colBase + 0)  * Kt + kbB;
  const _Float16* b1p = Bt + (size_t)(colBase + 16) * Kt + kbB;
  const _Float16* b2p = Bt + (size_t)(colBase + 32) * Kt + kbB;
  const _Float16* b3p = Bt + (size_t)(colBase + 48) * Kt + kbB;

  // staging role: 256 threads x 16B = 4KB, one b128 per 16-half segment of 32 rows x 128 halves
  int srow = threadIdx.x >> 3;   // 0..31
  int sseg = threadIdx.x & 7;    // 0..7, 16 halves each
  int rg = bM * 32 + srow;
  if (rg > M - 1) rg = M - 1;    // clamp (stores are guarded)
  const _Float16* gsrc = Ah + (size_t)rg * Kt + sseg * 16;
  _Float16* lbase = (_Float16*)As + srow * KB + sseg * 16;

  v8f c00 = {}, c01 = {}, c02 = {}, c03 = {};
  v8f c10 = {}, c11 = {}, c12 = {}, c13 = {};

  // issue first stage
  {
    unsigned laddr = (unsigned)(uintptr_t)(const void*)lbase;
    unsigned long long gaddr = (unsigned long long)(uintptr_t)(const void*)gsrc;
    asm volatile("global_load_async_to_lds_b128 %0, %1, off"
                 :: "v"(laddr), "v"(gaddr) : "memory");
  }
  int nKB = Kt / KB;
  for (int kb = 0; kb < nKB; ++kb) {
    if (kb + 1 < nKB) {  // prefetch next K-block into the other buffer
      unsigned laddr = (unsigned)(uintptr_t)(const void*)(lbase + ((kb + 1) & 1) * (32 * KB));
      unsigned long long gaddr =
          (unsigned long long)(uintptr_t)(const void*)(gsrc + (kb + 1) * KB);
      asm volatile("global_load_async_to_lds_b128 %0, %1, off"
                   :: "v"(laddr), "v"(gaddr) : "memory");
      asm volatile("s_wait_asynccnt 0x1" ::: "memory");
    } else {
      asm volatile("s_wait_asynccnt 0x0" ::: "memory");
    }
    __syncthreads();   // this buffer's A slab visible to all waves
    const _Float16* Ab = (const _Float16*)As + (kb & 1) * (32 * KB);
#pragma unroll
    for (int kk = 0; kk < KB; kk += 32) {
      const _Float16* r0 = Ab + (lane & 15) * KB + kk + kbA;
      const _Float16* r1 = Ab + (16 + (lane & 15)) * KB + kk + kbA;
      v8h a0lo = *(const v8h*)(r0);
      v8h a0hi = *(const v8h*)(r0 + 8);
      v8h a1lo = *(const v8h*)(r1);
      v8h a1hi = *(const v8h*)(r1 + 8);
      v16h a0, a1;
#pragma unroll
      for (int i = 0; i < 8; ++i) {
        a0[i] = a0lo[i]; a0[8 + i] = a0hi[i];
        a1[i] = a1lo[i]; a1[8 + i] = a1hi[i];
      }
      int kg = kb * KB + kk;
      v16h b0 = *(const v16h*)(b0p + kg);
      v16h b1 = *(const v16h*)(b1p + kg);
      v16h b2 = *(const v16h*)(b2p + kg);
      v16h b3 = *(const v16h*)(b3p + kg);
      c00 = __builtin_amdgcn_wmma_f32_16x16x32_f16(false, a0, false, b0, (short)0, c00, false, false);
      c01 = __builtin_amdgcn_wmma_f32_16x16x32_f16(false, a0, false, b1, (short)0, c01, false, false);
      c02 = __builtin_amdgcn_wmma_f32_16x16x32_f16(false, a0, false, b2, (short)0, c02, false, false);
      c03 = __builtin_amdgcn_wmma_f32_16x16x32_f16(false, a0, false, b3, (short)0, c03, false, false);
      c10 = __builtin_amdgcn_wmma_f32_16x16x32_f16(false, a1, false, b0, (short)0, c10, false, false);
      c11 = __builtin_amdgcn_wmma_f32_16x16x32_f16(false, a1, false, b1, (short)0, c11, false, false);
      c12 = __builtin_amdgcn_wmma_f32_16x16x32_f16(false, a1, false, b2, (short)0, c12, false, false);
      c13 = __builtin_amdgcn_wmma_f32_16x16x32_f16(false, a1, false, b3, (short)0, c13, false, false);
    }
    __syncthreads();   // safe to overwrite the buffer we just consumed
  }

  int mHi = (lane >= 16) ? 8 : 0;
  int nlo = lane & 15;
  int cb = bN * 512 + wv * 64;
#pragma unroll
  for (int i = 0; i < 8; ++i) {
    int r0 = bM * 32 + i + mHi;
    int r1 = r0 + 16;
    if (r0 < M) {
      size_t rb = (size_t)r0 * Nt + cb;
      C[rb + 0 + nlo] = c00[i]; C[rb + 16 + nlo] = c01[i];
      C[rb + 32 + nlo] = c02[i]; C[rb + 48 + nlo] = c03[i];
    }
    if (r1 < M) {
      size_t rb = (size_t)r1 * Nt + cb;
      C[rb + 0 + nlo] = c10[i]; C[rb + 16 + nlo] = c11[i];
      C[rb + 32 + nlo] = c12[i]; C[rb + 48 + nlo] = c13[i];
    }
  }
}

// ---------- GAT edge kernels ----------
__global__ void k_init_bias(float* __restrict__ out, const float* __restrict__ b,
                            long N, int C) {
  long i = (long)blockIdx.x * blockDim.x + threadIdx.x;
  if (i >= N * C) return;
  out[i] = b[i % C];
}

__global__ void k_gat_logits(const float* __restrict__ xl, const float* __restrict__ xr,
                             const float* __restrict__ ea, const int* __restrict__ src,
                             const int* __restrict__ dst, const float* __restrict__ We,
                             const float* __restrict__ att, float* __restrict__ logits,
                             unsigned* __restrict__ m, int E, int dout) {
  int i = blockIdx.x * blockDim.x + threadIdx.x;
  if (i >= E * HH) return;
  int e = i >> 3, h = i & 7;
  int s = src[e], t = dst[e];
  float eav = ea[e];
  int HDp = dout * HH;
  const float* a  = xl + (size_t)s * HDp + h * dout;
  const float* bb = xr + (size_t)t * HDp + h * dout;
  const float* w  = We + h * dout;
  const float* at = att + h * dout;
  float acc = 0.f;
  for (int d = 0; d < dout; ++d) {
    float v = a[d] + bb[d] + eav * w[d];
    v = (v > 0.f) ? v : 0.2f * v;
    acc = fmaf(at[d], v, acc);
  }
  logits[i] = acc;
  atomicMax(&m[(size_t)t * HH + h], fenc(acc));
}

__global__ void k_gat_exden(const float* __restrict__ logits, const int* __restrict__ dst,
                            const unsigned* __restrict__ m, float* __restrict__ ex,
                            float* __restrict__ den, int E) {
  int i = blockIdx.x * blockDim.x + threadIdx.x;
  if (i >= E * HH) return;
  int e = i >> 3, h = i & 7;
  int t = dst[e];
  float v = __expf(logits[i] - fdec(m[(size_t)t * HH + h]));
  ex[i] = v;
  atomicAdd(&den[(size_t)t * HH + h], v);
}

__global__ void k_gat_accum(const float* __restrict__ xl, const int* __restrict__ src,
                            const int* __restrict__ dst, const float* __restrict__ ex,
                            const float* __restrict__ den, float* __restrict__ out,
                            int E, int dout) {
  int i = blockIdx.x * blockDim.x + threadIdx.x;
  if (i >= E * HH) return;
  int e = i >> 3, h = i & 7;
  int s = src[e], t = dst[e];
  float alpha = ex[i] / (den[(size_t)t * HH + h] + 1e-16f);
  int HDp = dout * HH;
  const float* a = xl + (size_t)s * HDp + h * dout;
  float* o = out + (size_t)t * HDp + h * dout;
  for (int d = 0; d < dout; ++d) atomicAdd(&o[d], a[d] * alpha);
}

// ---------- BatchNorm over concat(pa[N,Ca] | pb[N,Ca]) ----------
__global__ void k_bn_stats(const float* __restrict__ pa, const float* __restrict__ pb,
                           float* __restrict__ mean, float* __restrict__ rstd,
                           int N, int Ca) {
  int c = blockIdx.x;
  __shared__ float ss[256], sq[256];
  const float* srcp = (c < Ca) ? (pa + c) : (pb + (c - Ca));
  float s = 0.f, q = 0.f;
  for (int n = threadIdx.x; n < N; n += blockDim.x) {
    float v = srcp[(size_t)n * Ca];
    s += v; q += v * v;
  }
  ss[threadIdx.x] = s; sq[threadIdx.x] = q;
  __syncthreads();
  for (int st = 128; st > 0; st >>= 1) {
    if ((int)threadIdx.x < st) {
      ss[threadIdx.x] += ss[threadIdx.x + st];
      sq[threadIdx.x] += sq[threadIdx.x + st];
    }
    __syncthreads();
  }
  if (threadIdx.x == 0) {
    float mu = ss[0] / N;
    float var = sq[0] / N - mu * mu;
    mean[c] = mu;
    rstd[c] = rsqrtf(var + 1e-5f);
  }
}

__global__ void k_bn_apply(const float* __restrict__ pa, const float* __restrict__ pb,
                           const float* __restrict__ mean, const float* __restrict__ rstd,
                           const float* __restrict__ g, const float* __restrict__ bb,
                           float* __restrict__ out, long N, int Ca) {
  int Ct = 2 * Ca;
  long i = (long)blockIdx.x * blockDim.x + threadIdx.x;
  if (i >= N * Ct) return;
  int c = (int)(i % Ct);
  long n = i / Ct;
  float v = (c < Ca) ? pa[n * Ca + c] : pb[n * Ca + (c - Ca)];
  out[i] = (v - mean[c]) * rstd[c] * g[c] + bb[c];
}

// ---------- routes ----------
__global__ void k_route_accum(const float* __restrict__ emb, const int* __restrict__ rid,
                              float* __restrict__ rs, long N, int C) {
  long i = (long)blockIdx.x * blockDim.x + threadIdx.x;
  if (i >= N * C) return;
  int c = (int)(i % C);
  long n = i / C;
  atomicAdd(&rs[(size_t)rid[n] * C + c], emb[i]);
}

__global__ void k_route_cumsum(const float* __restrict__ rs, float* __restrict__ ptop,
                               int base) {
  int c = blockIdx.x * blockDim.x + threadIdx.x;
  if (c >= 1024) return;
  float run = 0.f;
  for (int r = 0; r < 64; ++r) {
    run += rs[(size_t)r * 1024 + c];
    ptop[(size_t)r * 4096 + base + c] = run;
  }
  float tot = run;
  for (int r = 0; r < 64; ++r)
    ptop[(size_t)r * 4096 + base + 1024 + c] =
        tot - ptop[(size_t)r * 4096 + base + c];
}

// ---------- small BN over [64,4096] + MLP ----------
__global__ void k_pbn(const float* __restrict__ ptop, const float* __restrict__ g,
                      const float* __restrict__ b, float* __restrict__ out) {
  int c = blockIdx.x * blockDim.x + threadIdx.x;
  if (c >= 4096) return;
  float s = 0.f, q = 0.f;
  for (int r = 0; r < 64; ++r) {
    float v = ptop[(size_t)r * 4096 + c];
    s += v; q += v * v;
  }
  float mu = s / 64.f;
  float var = q / 64.f - mu * mu;
  float rst = rsqrtf(var + 1e-5f);
  for (int r = 0; r < 64; ++r) {
    float v = ptop[(size_t)r * 4096 + c];
    out[(size_t)r * 4096 + c] = (v - mu) * rst * g[c] + b[c];
  }
}

__global__ void k_mlp(const float* __restrict__ in, const float* __restrict__ Wm,
                      const float* __restrict__ bias, float* __restrict__ out,
                      int Ni, int No, int mode) {
  int i = blockIdx.x * blockDim.x + threadIdx.x;
  if (i >= 64 * No) return;
  int r = i / No, j = i % No;
  float acc = bias[j];
  const float* ir = in + (size_t)r * Ni;
  for (int k = 0; k < Ni; ++k) acc = fmaf(ir[k], Wm[(size_t)k * No + j], acc);
  if (mode == 0) acc = (acc > 0.f) ? acc : 0.01f * acc;
  else if (mode == 2) acc = 1.0f / (1.0f + __expf(-acc));
  out[i] = acc;
}

// =====================================================================
extern "C" void kernel_launch(void* const* d_in, const int* in_sizes, int n_in,
                              void* d_out, int out_size, void* d_ws, size_t ws_size,
                              hipStream_t stream) {
  (void)in_sizes; (void)n_in; (void)out_size; (void)ws_size;
  const long N = NCLI;
  const float* x_p1    = (const float*)d_in[0];
  const float* x_p2    = (const float*)d_in[1];
  const float* x_full  = (const float*)d_in[2];
  const float* ea_p1   = (const float*)d_in[3];
  const float* ea_p2   = (const float*)d_in[4];
  const float* ea_full = (const float*)d_in[5];
  const int*   ei_p1   = (const int*)d_in[6];
  const int*   ei_p2   = (const int*)d_in[7];
  const int*   ei_full = (const int*)d_in[8];
  const int*   route_p1 = (const int*)d_in[9];
  const int*   route_p2 = (const int*)d_in[10];
  const float* Wl1 = (const float*)d_in[11], *Wr1 = (const float*)d_in[12];
  const float* We1 = (const float*)d_in[13], *att1 = (const float*)d_in[14];
  const float* b1  = (const float*)d_in[15];
  const float* Wl2 = (const float*)d_in[16], *Wr2 = (const float*)d_in[17];
  const float* We2 = (const float*)d_in[18], *att2 = (const float*)d_in[19];
  const float* b2  = (const float*)d_in[20];
  const float* Wl3 = (const float*)d_in[21], *Wr3 = (const float*)d_in[22];
  const float* We3 = (const float*)d_in[23], *att3 = (const float*)d_in[24];
  const float* b3  = (const float*)d_in[25];
  const float* bng = (const float*)d_in[26], *bnb = (const float*)d_in[27];
  const float* png = (const float*)d_in[28], *pnb = (const float*)d_in[29];
  const float* Ws[8], *cs[8];
  for (int l = 0; l < 8; ++l) { Ws[l] = (const float*)d_in[30 + 2 * l]; cs[l] = (const float*)d_in[31 + 2 * l]; }
  const int mdims[9] = {4096, 2048, 1024, 512, 256, 128, 64, 32, 1};

  // ---- carve workspace ----
  char* w = (char*)d_ws;
  auto carve = [&](size_t bytes) -> void* {
    void* p = (void*)w;
    w += (bytes + 255) & ~(size_t)255;
    return p;
  };
  _Float16* WT1l = (_Float16*)carve((size_t)512 * 32 * 2);
  _Float16* WT1r = (_Float16*)carve((size_t)512 * 32 * 2);
  _Float16* WT2l = (_Float16*)carve((size_t)512 * 32 * 2);
  _Float16* WT2r = (_Float16*)carve((size_t)512 * 32 * 2);
  _Float16* WT3l = (_Float16*)carve((size_t)1024 * 1024 * 2);
  _Float16* WT3r = (_Float16*)carve((size_t)1024 * 1024 * 2);
  _Float16* XH   = (_Float16*)carve((size_t)N * 1024 * 2);
  float* A1  = (float*)carve((size_t)N * 1024 * 4);
  float* B1  = (float*)carve((size_t)N * 1024 * 4);
  float* C1  = (float*)carve((size_t)N * 1024 * 4);
  float* FG  = (float*)carve((size_t)N * 512 * 4);
  float* PE  = (float*)carve((size_t)N * 512 * 4);
  float* LG  = (float*)carve((size_t)EFE * 8 * 4);
  float* EXB = (float*)carve((size_t)EFE * 8 * 4);
  unsigned* Mb = (unsigned*)carve((size_t)N * 8 * 4);
  float* DN  = (float*)carve((size_t)N * 8 * 4);
  float* MEAN = (float*)carve(4096 * 4);
  float* RSTD = (float*)carve(4096 * 4);
  float* RS   = (float*)carve((size_t)64 * 1024 * 4);
  float* PTOP = (float*)carve((size_t)64 * 4096 * 4);
  float* H0   = (float*)carve((size_t)64 * 4096 * 4);
  float* H1   = (float*)carve((size_t)64 * 4096 * 4);

  auto nb = [](long n, int t) { return (unsigned)((n + t - 1) / t); };
  auto gemm = [&](const _Float16* A, const _Float16* B, float* C, int M, int Nt, int Kt) {
    long waves = (long)(M / 16) * (Nt / 64);
    k_wmma_gemm<<<dim3(nb(waves, 8)), dim3(256), 0, stream>>>(A, B, C, M, Nt, Kt);
  };
  auto gemm_lds = [&](const _Float16* A, const _Float16* B, float* C, int M, int Nt, int Kt) {
    int blocksN = Nt >> 9;
    int blocksM = (M + 31) / 32;
    k_wmma_gemm_lds<<<dim3((unsigned)(blocksM * blocksN)), dim3(256), 0, stream>>>(A, B, C, M, Nt, Kt);
  };

  // ---- weight convert/transpose (f32 -> f16, [K,N] -> [N,Kp]) ----
  k_wT_f16<<<nb(512 * 32, 256), 256, 0, stream>>>(Wl1, WT1l, 16, 512, 32);
  k_wT_f16<<<nb(512 * 32, 256), 256, 0, stream>>>(Wr1, WT1r, 16, 512, 32);
  k_wT_f16<<<nb(512 * 32, 256), 256, 0, stream>>>(Wl2, WT2l, 16, 512, 32);
  k_wT_f16<<<nb(512 * 32, 256), 256, 0, stream>>>(Wr2, WT2r, 16, 512, 32);
  k_wT_f16<<<nb((long)1024 * 1024, 256), 256, 0, stream>>>(Wl3, WT3l, 1024, 1024, 1024);
  k_wT_f16<<<nb((long)1024 * 1024, 256), 256, 0, stream>>>(Wr3, WT3r, 1024, 1024, 1024);

  // ---- full graph: GATv2 layer (tag 2) -> FG[N,512] ----
  k_pad_f16<<<nb(N * 32, 256), 256, 0, stream>>>(x_full, XH, N, 16, 32);
  gemm(XH, WT2l, A1, (int)N, 512, 32);
  gemm(XH, WT2r, B1, (int)N, 512, 32);
  k_fill_u32<<<nb(N * 8, 256), 256, 0, stream>>>(Mb, 0u, N * 8);
  k_fill_u32<<<nb(N * 8, 256), 256, 0, stream>>>((unsigned*)DN, 0u, N * 8);
  k_gat_logits<<<nb((long)EFE * 8, 256), 256, 0, stream>>>(
      A1, B1, ea_full, ei_full, ei_full + EFE, We2, att2, LG, Mb, EFE, 64);
  k_gat_exden<<<nb((long)EFE * 8, 256), 256, 0, stream>>>(LG, ei_full + EFE, Mb, EXB, DN, EFE);
  k_init_bias<<<nb(N * 512, 256), 256, 0, stream>>>(FG, b2, N, 512);
  k_gat_accum<<<nb((long)EFE * 8, 256), 256, 0, stream>>>(
      A1, ei_full, ei_full + EFE, EXB, DN, FG, EFE, 64);

  // ---- per-parent pipeline ----
  auto run_parent = [&](const float* xp, const float* eap, const int* eip,
                        const int* route, int base) {
    // layer 1 (tag 1): -> PE[N,512]
    k_pad_f16<<<nb(N * 32, 256), 256, 0, stream>>>(xp, XH, N, 16, 32);
    gemm(XH, WT1l, A1, (int)N, 512, 32);
    gemm(XH, WT1r, B1, (int)N, 512, 32);
    k_fill_u32<<<nb(N * 8, 256), 256, 0, stream>>>(Mb, 0u, N * 8);
    k_fill_u32<<<nb(N * 8, 256), 256, 0, stream>>>((unsigned*)DN, 0u, N * 8);
    k_gat_logits<<<nb((long)EPE * 8, 256), 256, 0, stream>>>(
        A1, B1, eap, eip, eip + EPE, We1, att1, LG, Mb, EPE, 64);
    k_gat_exden<<<nb((long)EPE * 8, 256), 256, 0, stream>>>(LG, eip + EPE, Mb, EXB, DN, EPE);
    k_init_bias<<<nb(N * 512, 256), 256, 0, stream>>>(PE, b1, N, 512);
    k_gat_accum<<<nb((long)EPE * 8, 256), 256, 0, stream>>>(
        A1, eip, eip + EPE, EXB, DN, PE, EPE, 64);
    // concat + BN -> C1[N,1024]
    k_bn_stats<<<1024, 256, 0, stream>>>(PE, FG, MEAN, RSTD, (int)N, 512);
    k_bn_apply<<<nb(N * 1024, 256), 256, 0, stream>>>(PE, FG, MEAN, RSTD, bng, bnb, C1, N, 512);
    // layer 3 (tag 3): big WMMA GEMMs, K=1024, LDS-pipelined with async loads
    k_pad_f16<<<nb(N * 1024, 256), 256, 0, stream>>>(C1, XH, N, 1024, 1024);
    gemm_lds(XH, WT3l, A1, (int)N, 1024, 1024);
    gemm_lds(XH, WT3r, B1, (int)N, 1024, 1024);
    k_fill_u32<<<nb(N * 8, 256), 256, 0, stream>>>(Mb, 0u, N * 8);
    k_fill_u32<<<nb(N * 8, 256), 256, 0, stream>>>((unsigned*)DN, 0u, N * 8);
    k_gat_logits<<<nb((long)EPE * 8, 256), 256, 0, stream>>>(
        A1, B1, eap, eip, eip + EPE, We3, att3, LG, Mb, EPE, 128);
    k_gat_exden<<<nb((long)EPE * 8, 256), 256, 0, stream>>>(LG, eip + EPE, Mb, EXB, DN, EPE);
    k_init_bias<<<nb(N * 1024, 256), 256, 0, stream>>>(C1, b3, N, 1024);
    k_gat_accum<<<nb((long)EPE * 8, 256), 256, 0, stream>>>(
        A1, eip, eip + EPE, EXB, DN, C1, EPE, 128);
    // route sums + prefix sums into PTOP
    k_fill_u32<<<nb((long)64 * 1024, 256), 256, 0, stream>>>((unsigned*)RS, 0u, (long)64 * 1024);
    k_route_accum<<<nb(N * 1024, 256), 256, 0, stream>>>(C1, route, RS, N, 1024);
    k_route_cumsum<<<4, 256, 0, stream>>>(RS, PTOP, base);
  };

  run_parent(x_p1, ea_p1, ei_p1, route_p1, 0);
  run_parent(x_p2, ea_p2, ei_p2, route_p2, 2048);

  // ---- top BN + MLP head ----
  k_pbn<<<16, 256, 0, stream>>>(PTOP, png, pnb, H0);
  float* cur = H0;
  float* nxt = H1;
  for (int l = 0; l < 8; ++l) {
    int mode = (l == 7) ? 2 : 0;
    float* outp = (l == 7) ? (float*)d_out : nxt;
    k_mlp<<<nb((long)64 * mdims[l + 1], 256), 256, 0, stream>>>(
        cur, Ws[l], cs[l], outp, mdims[l], mdims[l + 1], mode);
    float* t = cur; cur = nxt; nxt = t;
  }
  // ptop_batch = zeros(64, int32) -> second half of d_out
  k_fill_u32<<<1, 64, 0, stream>>>(((unsigned*)d_out) + 64, 0u, 64);
}
